// Downsample_PASA_group_softmax_59081570124042
// MI455X (gfx1250) — matrix operand order repaired and checked by reading, not compiled, not run on hardware
//
#include <hip/hip_runtime.h>
#include <hip/hip_bf16.h>

typedef __attribute__((ext_vector_type(16))) __bf16 v16bf;
typedef __attribute__((ext_vector_type(8)))  float  v8f;

#define C_IN    256
#define HW_IN   128
#define OHW     64
#define CO      72
#define CO_P    80
#define NMT     5          // M-tiles of 16 (80 padded channels)
#define KDIM    2304       // 256 * 9
#define NKT     72         // K-tiles of 32
#define KHALF   36         // K-tiles per wave-group
#define NPT     4          // position (N) tiles of 16
#define TILE_OH 8
#define TILE_OW 8
#define NPOS    64         // positions per workgroup tile
#define NTHR    256        // 8 waves
#define BN_EPS  1e-5f

#define WFRAG_ELEMS (NKT * NMT * 32 * 16)          // 184320 bf16
#define WFRAG_BYTES (WFRAG_ELEMS * 2)              // 368640 B
#define IM2_ELEMS   (KDIM * NPOS)                  // 147456
#define IM2_BYTES   (IM2_ELEMS * 2)                // 294912 B
#define SG_BYTES    (CO_P * NPOS * 4)              // 20480 B

__device__ __forceinline__ int reflect_idx(int i) {
  if (i < 0) i = -i;
  if (i > HW_IN - 1) i = 2 * (HW_IN - 1) - i;
  return i;
}

// LDS im2col in WMMA B-fragment order, this code's contiguous k order (matches A):
// element (k, p):  kt=k/32, hsel=(k%32)/16, i=k%16, pt=p/16, n=p%16
// addr = (((kt*NPT + pt)*32) + hsel*16 + n)*16 + i   (32B contiguous per lane)
__device__ __forceinline__ int frag_addr(int k, int p) {
  int kt = k >> 5, koff = k & 31;
  int hsel = koff >> 4, i = koff & 15;
  int pt = p >> 4, n = p & 15;
  return (((kt * NPT + pt) * 32) + hsel * 16 + n) * 16 + i;
}

// ---------------------------------------------------------------------------
// Prep: fold BN into conv weights, bf16-ize, swizzle into WMMA A-fragment
// order (16x32 bf16): lane = 16*half + m, K = 16*half + i (this code's contiguous k order, same as B)
// ---------------------------------------------------------------------------
__global__ void pasa_prep(const float* __restrict__ w, const float* __restrict__ gamma,
                          const float* __restrict__ beta, const float* __restrict__ mean,
                          const float* __restrict__ var,
                          __bf16* __restrict__ wfrag, float* __restrict__ bias) {
  int idx = blockIdx.x * blockDim.x + threadIdx.x;
  if (idx >= CO_P * KDIM) return;
  int co = idx / KDIM, k = idx - co * KDIM;
  float s = 0.f, wv = 0.f;
  if (co < CO) {
    s  = gamma[co] * rsqrtf(var[co] + BN_EPS);
    wv = w[(size_t)co * KDIM + k] * s;
  }
  int mt = co >> 4, m = co & 15;
  int kt = k >> 5,  koff = k & 31;
  int hsel = koff >> 4, i = koff & 15;
  int lane = hsel * 16 + m;
  wfrag[((size_t)(kt * NMT + mt) * 32 + lane) * 16 + i] = (__bf16)wv;
  if (k == 0) bias[co] = (co < CO) ? (beta[co] - mean[co] * s) : 0.f;
}

// ---------------------------------------------------------------------------
// Main, per 8x8 strided-position tile (256 threads = 8 waves):
//   phase 0: zero sigma buffer; build bf16 im2col (2304x64) in LDS, frag order
//   phase 1: sigma = W(80x2304) @ im2col via v_wmma_f32_16x16x32_bf16
//            8 waves: wave w -> N-tile w&3, K-half w>>2; double-buffered
//            A/B fragments; partials combined with ds_add_f32
//   phase 1b: per-position softmax over 72 channels (64 threads, LDS)
//   phase 2: 9-tap adaptive weighted sum from im2col LDS, strided store
// ---------------------------------------------------------------------------
__global__ void __launch_bounds__(NTHR)
pasa_main(const float* __restrict__ x, const __bf16* __restrict__ wfrag,
          const float* __restrict__ bias, float* __restrict__ out) {
  extern __shared__ char smem[];
  __bf16* im2 = (__bf16*)smem;                     // im2col, frag order
  float*  sg  = (float*)(smem + IM2_BYTES);        // [CO_P][NPOS] sigma

  const int tid = threadIdx.x;
  const int blk = blockIdx.x;
  const int b   = blk >> 6;                        // batch
  const int t   = blk & 63;
  const int oh0 = (t >> 3) * TILE_OH;
  const int ow0 = (t & 7) * TILE_OW;

  const float* xb = x + (size_t)b * C_IN * HW_IN * HW_IN;

  // ---- phase 0: zero sigma accumulator, build im2col in LDS ----
  for (int idx = tid; idx < CO_P * NPOS; idx += NTHR) sg[idx] = 0.f;
  for (int idx = tid; idx < KDIM * NPOS; idx += NTHR) {
    int k = idx >> 6;
    int p = idx & 63;
    int c  = k / 9;
    int tp = k - c * 9;
    int kh = tp / 3;
    int kw = tp - kh * 3;
    int doh = p >> 3, dow = p & 7;
    int gr = reflect_idx(2 * (oh0 + doh) + kh - 1);
    int gc = reflect_idx(2 * (ow0 + dow) + kw - 1);
    float v = xb[(size_t)(c * HW_IN + gr) * HW_IN + gc];
    im2[frag_addr(k, p)] = (__bf16)v;
  }
  __syncthreads();

  // ---- phase 1: WMMA GEMM, double-buffered fragments ----
  const int lane  = tid & 31;
  const int wave  = tid >> 5;                      // 0..7
  const int ntile = wave & 3;                      // N-tile (16 positions)
  const int khalf = wave >> 2;                     // K range half
  const int hsel  = lane >> 4;
  const int n     = lane & 15;
  const int p     = ntile * 16 + n;
  const int kt0   = khalf * KHALF;

  v8f acc[NMT] = {};
  const v16bf* wf = (const v16bf*)wfrag;
  const v16bf* bl = (const v16bf*)im2;             // LDS as 32B fragments

  v16bf bf_cur = bl[(kt0 * NPT + ntile) * 32 + lane];
  v16bf af_cur[NMT];
#pragma unroll
  for (int mt = 0; mt < NMT; ++mt)
    af_cur[mt] = wf[(size_t)(kt0 * NMT + mt) * 32 + lane];

  for (int kk = 0; kk < KHALF - 1; ++kk) {
    int ktn = kt0 + kk + 1;
    v16bf bf_nxt = bl[(ktn * NPT + ntile) * 32 + lane];
    v16bf af_nxt[NMT];
#pragma unroll
    for (int mt = 0; mt < NMT; ++mt)
      af_nxt[mt] = wf[(size_t)(ktn * NMT + mt) * 32 + lane];
#pragma unroll
    for (int mt = 0; mt < NMT; ++mt)
      acc[mt] = __builtin_amdgcn_wmma_f32_16x16x32_bf16(
          false, af_cur[mt], false, bf_cur, (short)0, acc[mt], false, false);
    bf_cur = bf_nxt;
#pragma unroll
    for (int mt = 0; mt < NMT; ++mt) af_cur[mt] = af_nxt[mt];
  }
#pragma unroll
  for (int mt = 0; mt < NMT; ++mt)
    acc[mt] = __builtin_amdgcn_wmma_f32_16x16x32_bf16(
        false, af_cur[mt], false, bf_cur, (short)0, acc[mt], false, false);

  // combine K-half partials: ds_add_f32 into sigma buffer
  // C/D layout: VGPR j of tile mt, lane = hsel*16+n holds channel 16*mt+8*hsel+j
#pragma unroll
  for (int mt = 0; mt < NMT; ++mt)
#pragma unroll
    for (int j = 0; j < 8; ++j) {
      int m = mt * 16 + hsel * 8 + j;
      if (m < CO) atomicAdd(&sg[m * NPOS + p], acc[mt][j]);
    }
  __syncthreads();

  // ---- phase 1b: softmax over 72 channels, one thread per position ----
  if (tid < NPOS) {
    float vals[CO];
    float mx = -3.4e38f;
#pragma unroll 1
    for (int m = 0; m < CO; ++m) {
      float v = sg[m * NPOS + tid] + bias[m];
      vals[m] = v;
      mx = fmaxf(mx, v);
    }
    float sum = 0.f;
#pragma unroll 1
    for (int m = 0; m < CO; ++m) {
      float e = __expf(vals[m] - mx);
      vals[m] = e;
      sum += e;
    }
    float inv = 1.f / sum;
#pragma unroll 1
    for (int m = 0; m < CO; ++m) sg[m * NPOS + tid] = vals[m] * inv;
  }
  __syncthreads();

  // ---- phase 2: adaptive 9-tap weighted sum from im2col LDS ----
  const int p2  = tid & 63;
  const int sub = tid >> 6;                        // 0..3 -> channel quarters
  const int d2h = p2 >> 3, d2w = p2 & 7;
  const int oh = oh0 + d2h, ow = ow0 + d2w;
  float* ob = out + (size_t)b * C_IN * OHW * OHW;
#pragma unroll 1
  for (int g = 0; g < 8; ++g) {
    float s9[9];
#pragma unroll
    for (int tp = 0; tp < 9; ++tp) s9[tp] = sg[(g * 9 + tp) * NPOS + p2];
#pragma unroll 1
    for (int cl = 0; cl < 8; ++cl) {
      int c = g * 32 + sub * 8 + cl;
      float a2 = 0.f;
#pragma unroll
      for (int tp = 0; tp < 9; ++tp) {
        a2 += (float)im2[frag_addr(c * 9 + tp, p2)] * s9[tp];
      }
      ob[(size_t)(c * OHW + oh) * OHW + ow] = a2;
    }
  }
}

extern "C" void kernel_launch(void* const* d_in, const int* in_sizes, int n_in,
                              void* d_out, int out_size, void* d_ws, size_t ws_size,
                              hipStream_t stream) {
  (void)in_sizes; (void)n_in; (void)out_size; (void)ws_size;
  const float* x     = (const float*)d_in[0];
  const float* convw = (const float*)d_in[1];
  const float* gamma = (const float*)d_in[2];
  const float* beta  = (const float*)d_in[3];
  const float* mean  = (const float*)d_in[4];
  const float* var   = (const float*)d_in[5];
  float* out = (float*)d_out;

  __bf16* wfrag = (__bf16*)d_ws;
  float*  bias  = (float*)((char*)d_ws + WFRAG_BYTES);

  // fold BN + swizzle weights into WMMA fragment order
  pasa_prep<<<(CO_P * KDIM + 255) / 256, 256, 0, stream>>>(
      convw, gamma, beta, mean, var, wfrag, bias);

  // 4 batches * 64 tiles, 256 threads (8 waves), ~308 KB dynamic LDS
  size_t smem = (size_t)IM2_BYTES + SG_BYTES;
  pasa_main<<<dim3(4 * 64), dim3(NTHR), smem, stream>>>(x, wfrag, bias, out);
}
